// PolyHashV2_42606075576716
// MI455X (gfx1250) — matrix operand order repaired and checked by reading, not compiled, not run on hardware
//
#include <hip/hip_runtime.h>
#include <stdint.h>

// ---------------------------------------------------------------------------
// PolyHash MLP for MI455X (gfx1250): bf16 WMMA GEMM chain.
//   feats(bf16)[32768,1152] -> h = feats@W1+b (wmma) -> 2x (relu+res+LN, wmma)
//   -> out = h@Wo+bo (wmma, f32 out)
// All matrix math on v_wmma_f32_16x16x32_bf16, f32 accumulation.
// GEMM uses 64x64 register-blocked wave tiles: 16 WMMAs per 16 b128 loads.
// ---------------------------------------------------------------------------

typedef __attribute__((ext_vector_type(16))) __bf16 v16bf;
typedef __attribute__((ext_vector_type(8)))  float v8f;

#define BATCH      8
#define SEQ        4096
#define NPOS       (BATCH * SEQ)          // 32768 rows
#define VOCAB      1024
#define BYTE_DIM   128
#define NUM_TABLES 32
#define EMBED      32
#define FEAT       (BYTE_DIM + NUM_TABLES * EMBED)  // 1152
#define HIDDEN     512
#define BUCKETS    131072                 // 2^17 -> hash mod == low-17-bit mask
#define BUCKET_MASK (BUCKETS - 1)

// HASH_PRIMES (48) — only low 32 bits of tok*prime matter since BUCKETS=2^17.
__constant__ uint32_t c_primes[48] = {
  2654435761u, 2246822519u, 3266489917u, 2028178513u, 1220703125u, 1610612741u,
  805306457u,  402653189u,  3674653429u, 2860486313u, 1073676287u, 2971215073u,
  1500450271u, 3267000013u, 2654435789u, 4049292737u, 2246822531u, 3266489927u,
  2028178519u, 1220703133u, 1610612743u, 805306459u,  402653191u,  3674653433u,
  2654435771u, 2246822527u, 3266489933u, 2028178529u, 1220703137u, 1610612747u,
  805306463u,  402653197u,  3674653441u, 2860486319u, 1073676293u, 2971215077u,
  1500450281u, 3267000017u, 2654435801u, 4049292743u, 2246822537u, 3266489939u,
  2028178531u, 1220703143u, 1610612753u, 805306467u,  402653201u,  3674653447u };

// _generate_patterns(32, 8) from the reference, expanded.
__constant__ uint8_t c_pat_off[NUM_TABLES][3] = {
  {1,0,0},{2,0,0},{3,0,0},{4,0,0},{5,0,0},{6,0,0},{7,0,0},{8,0,0},
  {1,2,0},{2,3,0},{3,4,0},{1,3,0},{2,4,0},{1,4,0},{1,5,0},{2,5,0},
  {3,5,0},{1,6,0},{2,6,0},{1,7,0},
  {1,2,3},{1,2,4},{1,3,5},{2,3,4},
  {1,8,0},{1,9,0},{1,10,0},{1,11,0},{1,12,0},{1,13,0},{1,14,0},{1,15,0} };
__constant__ uint8_t c_pat_len[NUM_TABLES] = {
  1,1,1,1,1,1,1,1, 2,2,2,2,2,2,2,2, 2,2,2,2, 3,3,3,3, 2,2,2,2,2,2,2,2 };

__device__ __forceinline__ uint16_t f32_to_bf16(float f) {
  uint32_t u = __float_as_uint(f);
  u += 0x7FFFu + ((u >> 16) & 1u);   // RNE
  return (uint16_t)(u >> 16);
}

// ---------------------------------------------------------------------------
// 1) Hash + gather + byte-embed -> bf16 feature matrix [NPOS, FEAT].
//    One wave per position; lane t owns hash table t.
// ---------------------------------------------------------------------------
__global__ __launch_bounds__(256) void feats_kernel(
    const long long* __restrict__ tokens,      // [NPOS] int64
    const float*     __restrict__ byte_embed,  // [VOCAB, BYTE_DIM]
    const float*     __restrict__ hash_tables, // [T, BUCKETS, EMBED]
    uint16_t*        __restrict__ feats)       // [NPOS, FEAT] bf16
{
  const int lane = threadIdx.x & 31;
  const int row  = blockIdx.x * (blockDim.x >> 5) + (threadIdx.x >> 5);
  if (row >= NPOS) return;
  const int s = row & (SEQ - 1);

  // Polynomial XOR hash for table t = lane (32-bit math is exact mod 2^17).
  uint32_t h = 0;
  const int len = c_pat_len[lane];
  #pragma unroll 3
  for (int k = 0; k < 3; ++k) {
    if (k < len) {
      const int off = c_pat_off[lane][k];
      uint32_t tok = (s >= off) ? (uint32_t)tokens[row - off] : 0u;
      h ^= tok * c_primes[(lane * 3 + k) % 48];
    }
  }
  const uint32_t bidx = h & BUCKET_MASK;

  const size_t fbase = (size_t)row * FEAT;
  // Byte embedding (128 floats, 4 per lane).
  const uint32_t tok0 = (uint32_t)tokens[row];
  #pragma unroll
  for (int j = lane; j < BYTE_DIM; j += 32)
    feats[fbase + j] = f32_to_bf16(byte_embed[(size_t)tok0 * BYTE_DIM + j]);
  // Hash features: broadcast bucket of table t, coalesced 128B row gather.
  #pragma unroll 4
  for (int t = 0; t < NUM_TABLES; ++t) {
    const uint32_t bt = (uint32_t)__shfl((int)bidx, t, 32);
    const float v = hash_tables[((size_t)t * BUCKETS + bt) * EMBED + lane];
    feats[fbase + BYTE_DIM + t * EMBED + lane] = f32_to_bf16(v);
  }
}

// ---------------------------------------------------------------------------
// 2) Weight transpose + f32->bf16: WT[n*K+k] = bf16(W[k*N+n]).
// ---------------------------------------------------------------------------
__global__ __launch_bounds__(256) void transpose_bf16_kernel(
    const float* __restrict__ W, uint16_t* __restrict__ WT, int K, int N)
{
  const int idx = blockIdx.x * blockDim.x + threadIdx.x;
  if (idx >= K * N) return;
  const int n = idx / K, k = idx - n * K;
  WT[idx] = f32_to_bf16(W[(size_t)k * N + n]);
}

// ---------------------------------------------------------------------------
// 3) Register-blocked bf16 WMMA GEMM: C[M,N] = A[M,K] @ WT[N,K]^T + bias.
//    Each wave owns a 64(M) x 64(N) tile: 4 A frags x 4 B frags ->
//    16 f32 accumulators (128 VGPRs), 16 WMMAs per K-step of 32 for only
//    16 b128 loads (1.0 load/wmma).  All loads issue before the WMMA burst
//    so one partial s_wait covers the step instead of one wait per WMMA.
//    Fragment loads follow the CDNA5 ISA VGPR layouts:
//      A 16x32 bf16: lanes 0-15 (M=lane) K chunks {k0..k0+7, k0+16..k0+23},
//                    lanes 16-31 K chunks {k0+8..15, k0+24..31}  -> 2x b128.
//      B 32x16 bf16: lanes 0-15 (N=lane) K chunks {k0..k0+7, k0+16..k0+23},
//                    lanes 16-31 K chunks {k0+8..15, k0+24..31}  -> 2x b128.
// ---------------------------------------------------------------------------
union FragBF { v16bf v; uint4 q[2]; };

__global__ __launch_bounds__(256) void gemm_bf16_kernel(
    const uint16_t* __restrict__ A,     // [M,K] bf16 row-major
    const uint16_t* __restrict__ WT,    // [N,K] bf16 (pre-transposed weight)
    const float*    __restrict__ bias,  // [N]
    float*          __restrict__ outF,  // [M,N] f32
    uint16_t*       __restrict__ outH,  // [M,N] bf16 (optional, may be null)
    int M, int N, int K)
{
  const int lane     = threadIdx.x & 31;
  const int wave     = blockIdx.x * (blockDim.x >> 5) + (threadIdx.x >> 5);
  const int ntiles64 = N >> 6;
  const int mtile    = wave / ntiles64;   // 64-row band
  const int nband    = wave % ntiles64;   // 64-col band
  if (mtile * 64 >= M) return;            // uniform per-wave: EXEC stays all-1s

  const int hi   = lane >> 4;             // lane half select
  const int lo16 = lane & 15;
  const int m0   = mtile * 64;
  const int col0 = nband * 64;

  const uint16_t* Abase[4];
  const uint16_t* Bbase[4];
  #pragma unroll
  for (int i = 0; i < 4; ++i)
    Abase[i] = A  + (size_t)(m0   + i * 16 + lo16) * K + hi * 8;   // A interleave
  #pragma unroll
  for (int i = 0; i < 4; ++i)
    Bbase[i] = WT + (size_t)(col0 + i * 16 + lo16) * K + hi * 8;   // B interleave

  v8f acc[4][4] = {};                     // [mi][ni], zero-accumulate

  for (int k0 = 0; k0 < K; k0 += 32) {
    FragBF a[4], b[4];
    // Issue all 16 b128 loads up front (clause-friendly), then the WMMA burst.
    #pragma unroll
    for (int i = 0; i < 4; ++i) {
      a[i].q[0] = *(const uint4*)(Abase[i] + k0);
      a[i].q[1] = *(const uint4*)(Abase[i] + k0 + 16);
    }
    #pragma unroll
    for (int i = 0; i < 4; ++i) {
      b[i].q[0] = *(const uint4*)(Bbase[i] + k0);
      b[i].q[1] = *(const uint4*)(Bbase[i] + k0 + 16);
    }
    #pragma unroll
    for (int mi = 0; mi < 4; ++mi)
      #pragma unroll
      for (int ni = 0; ni < 4; ++ni)
        acc[mi][ni] = __builtin_amdgcn_wmma_f32_16x16x32_bf16(
            /*neg_a=*/false, a[mi].v, /*neg_b=*/false, b[ni].v,
            /*c_mod=*/(short)0, acc[mi][ni],
            /*reuse_a=*/false, /*reuse_b=*/false);
  }

  // C/D layout: lane owns col = col0+ni*16+lo16, rows r + hi*8 (r = VGPR idx).
  #pragma unroll
  for (int ni = 0; ni < 4; ++ni) {
    const int col = col0 + ni * 16 + lo16;
    const float bv = bias ? bias[col] : 0.0f;
    #pragma unroll
    for (int mi = 0; mi < 4; ++mi) {
      #pragma unroll
      for (int r = 0; r < 8; ++r) {
        const int m = m0 + mi * 16 + r + hi * 8;
        const float v = acc[mi][ni][r] + bv;
        outF[(size_t)m * N + col] = v;
        if (outH) outH[(size_t)m * N + col] = f32_to_bf16(v);
      }
    }
  }
}

// ---------------------------------------------------------------------------
// 4) Fused epilogue: hh = relu(pre) + h ; h' = LN(hh)*w + b.
//    One wave per row of 512; 16 elems/lane; shfl_xor reductions.
// ---------------------------------------------------------------------------
__global__ __launch_bounds__(256) void relu_res_ln_kernel(
    const float* __restrict__ pre, const float* __restrict__ hin,
    const float* __restrict__ lw,  const float* __restrict__ lb,
    float* __restrict__ hout, uint16_t* __restrict__ houtb, int M)
{
  const int lane = threadIdx.x & 31;
  const int row  = blockIdx.x * (blockDim.x >> 5) + (threadIdx.x >> 5);
  if (row >= M) return;
  const size_t base = (size_t)row * HIDDEN;

  float x[16], s = 0.0f, s2 = 0.0f;
  #pragma unroll
  for (int i = 0; i < 16; ++i) {
    const int j = lane + i * 32;
    const float p = pre[base + j];
    const float v = (p > 0.0f ? p : 0.0f) + hin[base + j];
    x[i] = v; s += v; s2 += v * v;
  }
  #pragma unroll
  for (int o = 16; o >= 1; o >>= 1) {
    s  += __shfl_xor(s,  o, 32);
    s2 += __shfl_xor(s2, o, 32);
  }
  const float mean = s * (1.0f / HIDDEN);
  const float var  = s2 * (1.0f / HIDDEN) - mean * mean;
  const float rstd = rsqrtf(var + 1e-5f);
  #pragma unroll
  for (int i = 0; i < 16; ++i) {
    const int j = lane + i * 32;
    const float v = (x[i] - mean) * rstd * lw[j] + lb[j];
    hout[base + j]  = v;
    houtb[base + j] = f32_to_bf16(v);
  }
}

// ---------------------------------------------------------------------------
// Host orchestration.
// ---------------------------------------------------------------------------
static inline size_t align256(size_t x) { return (x + 255) & ~(size_t)255; }

extern "C" void kernel_launch(void* const* d_in, const int* in_sizes, int n_in,
                              void* d_out, int out_size, void* d_ws, size_t ws_size,
                              hipStream_t stream) {
  (void)in_sizes; (void)n_in; (void)out_size; (void)ws_size;
  const long long* tokens      = (const long long*)d_in[0];
  const float*     byte_embed  = (const float*)d_in[1];
  const float*     hash_tables = (const float*)d_in[2];
  const float*     in_w        = (const float*)d_in[3];
  const float*     in_b        = (const float*)d_in[4];
  const float*     res_w       = (const float*)d_in[5];   // [2,512,512]
  const float*     res_b       = (const float*)d_in[6];   // [2,512]
  const float*     ln_w        = (const float*)d_in[7];
  const float*     ln_b        = (const float*)d_in[8];
  const float*     out_w       = (const float*)d_in[9];
  const float*     out_b       = (const float*)d_in[10];

  // Workspace carve-out (~280 MB); `pre` aliases feats (feats dead post-GEMM1).
  uint8_t* ws = (uint8_t*)d_ws;
  size_t off = 0;
  uint16_t* feats = (uint16_t*)(ws + off); off += align256((size_t)NPOS * FEAT * 2);
  uint16_t* wt_in = (uint16_t*)(ws + off); off += align256((size_t)FEAT * HIDDEN * 2);
  uint16_t* wt_r0 = (uint16_t*)(ws + off); off += align256((size_t)HIDDEN * HIDDEN * 2);
  uint16_t* wt_r1 = (uint16_t*)(ws + off); off += align256((size_t)HIDDEN * HIDDEN * 2);
  uint16_t* wt_out= (uint16_t*)(ws + off); off += align256((size_t)HIDDEN * VOCAB * 2);
  float*    hA_f  = (float*)   (ws + off); off += align256((size_t)NPOS * HIDDEN * 4);
  float*    hB_f  = (float*)   (ws + off); off += align256((size_t)NPOS * HIDDEN * 4);
  uint16_t* hA_b  = (uint16_t*)(ws + off); off += align256((size_t)NPOS * HIDDEN * 2);
  uint16_t* hB_b  = (uint16_t*)(ws + off); off += align256((size_t)NPOS * HIDDEN * 2);
  float*    pre   = (float*)feats;   // alias: 64MB <= 75.5MB, feats dead by then

  // Weight conversion (tiny).
  {
    int n1 = FEAT * HIDDEN;
    transpose_bf16_kernel<<<(n1 + 255) / 256, 256, 0, stream>>>(in_w, wt_in, FEAT, HIDDEN);
    int n2 = HIDDEN * HIDDEN;
    transpose_bf16_kernel<<<(n2 + 255) / 256, 256, 0, stream>>>(res_w,              wt_r0, HIDDEN, HIDDEN);
    transpose_bf16_kernel<<<(n2 + 255) / 256, 256, 0, stream>>>(res_w + (size_t)n2, wt_r1, HIDDEN, HIDDEN);
    int n3 = HIDDEN * VOCAB;
    transpose_bf16_kernel<<<(n3 + 255) / 256, 256, 0, stream>>>(out_w, wt_out, HIDDEN, VOCAB);
  }

  // Feature build: 8 waves/block -> 4096 blocks.
  feats_kernel<<<NPOS / 8, 256, 0, stream>>>(tokens, byte_embed, hash_tables, feats);

  // GEMM1: [32768,1152] @ [1152,512]; waves = (M/64)*(N/64) = 512*8.
  {
    int grid = (NPOS / 64) * (HIDDEN / 64) / 8;
    gemm_bf16_kernel<<<grid, 256, 0, stream>>>(feats, wt_in, in_b, hA_f, hA_b,
                                               NPOS, HIDDEN, FEAT);
  }

  // Residual layers.
  float*    hf = hA_f;  uint16_t* hb = hA_b;
  float*    gf = hB_f;  uint16_t* gb = hB_b;
  const uint16_t* wt_res[2] = { wt_r0, wt_r1 };
  for (int l = 0; l < 2; ++l) {
    int grid = (NPOS / 64) * (HIDDEN / 64) / 8;
    gemm_bf16_kernel<<<grid, 256, 0, stream>>>(hb, wt_res[l], res_b + l * HIDDEN,
                                               pre, (uint16_t*)nullptr,
                                               NPOS, HIDDEN, HIDDEN);
    relu_res_ln_kernel<<<NPOS / 8, 256, 0, stream>>>(pre, hf,
                                                     ln_w + l * HIDDEN, ln_b + l * HIDDEN,
                                                     gf, gb, NPOS);
    float* tf = hf; hf = gf; gf = tf;
    uint16_t* tb = hb; hb = gb; gb = tb;
  }

  // Output GEMM: [32768,512] @ [512,1024] -> d_out (f32).
  {
    int grid = (NPOS / 64) * (VOCAB / 64) / 8;
    gemm_bf16_kernel<<<grid, 256, 0, stream>>>(hb, wt_out, out_b,
                                               (float*)d_out, (uint16_t*)nullptr,
                                               NPOS, VOCAB, HIDDEN);
  }
}
